// PatchSmoothnessLoss_41231686042325
// MI455X (gfx1250) — compile-verified, matrix operand order checked by reading, not run on hardware
//
#include <hip/hip_runtime.h>

#define B_ 16
#define T_ 1024
#define D_ 256
#define NTILE (T_ / 16)   // 64 tiles along s / t

typedef __attribute__((ext_vector_type(16))) __bf16 v16bf;
typedef __attribute__((ext_vector_type(8)))  float  v8f;
typedef __attribute__((ext_vector_type(8)))  int    v8i;
typedef __attribute__((ext_vector_type(4)))  float  vf4;   // NT-loadable float4
typedef __attribute__((ext_vector_type(2)))  float  vf2;   // NT-loadable float2

// ---------------------------------------------------------------------------
// Prep: z (f32, row-major BxTxD) -> zb (bf16 RNE) ; z2[row] = sum_d z^2
// One block per row, 128 threads, 2 elements per thread (packed dword store).
// z is read once -> non-temporal load keeps L2 free for the hot zb buffer.
// ---------------------------------------------------------------------------
__device__ __forceinline__ unsigned int f32_to_bf16_rne(float f) {
    unsigned int u = __float_as_uint(f);
    return (u + 0x7fffu + ((u >> 16) & 1u)) >> 16;
}

__global__ __launch_bounds__(128)
void prep_kernel(const float* __restrict__ z,
                 unsigned int* __restrict__ zbu,   // packed pairs of bf16
                 float* __restrict__ z2) {
    const int row = blockIdx.x;          // 0 .. B*T-1
    const int d   = threadIdx.x;         // 0 .. 127
    const vf2 v = __builtin_nontemporal_load(
        (const vf2*)z + (size_t)row * (D_ / 2) + d);
    const unsigned int lo = f32_to_bf16_rne(v.x);
    const unsigned int hi = f32_to_bf16_rne(v.y);
    zbu[(size_t)row * (D_ / 2) + d] = lo | (hi << 16);

    __shared__ float red[128];
    red[d] = v.x * v.x + v.y * v.y;
    __syncthreads();
#pragma unroll
    for (int off = 64; off > 0; off >>= 1) {
        if (d < off) red[d] += red[d + off];
        __syncthreads();
    }
    if (d == 0) z2[row] = red[0];
}

// ---------------------------------------------------------------------------
// Fragment loader: 16-bit 16x32 A/B layout, lane holds row (M or N) = lane&15,
// K pairs K0 = ks*32 + 8*half + {0,2,4,6} and +16. Those are two contiguous
// 16-byte runs -> two global_load_b128 per K-step. These hit L2 (zb stays
// resident because the gt_dT stream is loaded non-temporally).
// ---------------------------------------------------------------------------
__device__ __forceinline__ v16bf load_frag(const uint4* __restrict__ zb4,
                                           size_t rowChunkBase, int ks, int half) {
    const uint4 q0 = zb4[rowChunkBase + (size_t)(ks * 4 + half)];
    const uint4 q1 = zb4[rowChunkBase + (size_t)(ks * 4 + 2 + half)];
    v8i ar;
    ar[0] = (int)q0.x; ar[1] = (int)q0.y; ar[2] = (int)q0.z; ar[3] = (int)q0.w;
    ar[4] = (int)q1.x; ar[5] = (int)q1.y; ar[6] = (int)q1.z; ar[7] = (int)q1.w;
    return __builtin_bit_cast(v16bf, ar);
}

// ---------------------------------------------------------------------------
// Main fused kernel: one block per (b, t-tile); 8 waves, each wave sweeps
// 8 s-tiles. Per tile: w from the non-temporal gt_dT stream (b128 + v_exp_f32),
// Gram tile via 8x v_wmma_f32_16x16x32_bf16, scalar accumulation.
// ---------------------------------------------------------------------------
__global__ __launch_bounds__(256)
void lap_energy_kernel(const vf4* __restrict__ gt4,           // gt_dT as float4
                       const uint4* __restrict__ zb4,         // bf16 z, 16B chunks
                       const float* __restrict__ z2,
                       float* __restrict__ accum) {           // [num(16) den(16)]
    const int b    = blockIdx.x >> 6;
    const int t0   = (blockIdx.x & 63) << 4;
    const int tid  = threadIdx.x;
    const int lane = tid & 31;
    const int wave = tid >> 5;
    const int ln   = lane & 15;
    const int half = lane >> 4;

    // exp(-x/(2 s^2)) = exp2(x * -1/(2 s^2 ln2))
    constexpr float LN2 = 0.69314718055994530942f;
    constexpr float CYX = -1.0f / (2.0f * 0.09f * 0.09f * LN2);
    constexpr float CHW = -1.0f / (2.0f * 0.30f * 0.30f * LN2);

    // A fragments: rows t0..t0+15 of zb, all 8 K-steps resident in VGPRs.
    v16bf afrag[8];
    {
        const size_t arowC = ((size_t)b * T_ + (t0 + ln)) * 32;  // 16B chunks/row
#pragma unroll
        for (int ks = 0; ks < 8; ++ks)
            afrag[ks] = load_frag(zb4, arowC, ks, half);
    }

    // C layout: vgpr j, lane -> (M = j + 8*half, N = lane&15)
    float z2t[8];
#pragma unroll
    for (int j = 0; j < 8; ++j) z2t[j] = z2[b * T_ + t0 + j + half * 8];

    float accN = 0.0f, accD = 0.0f;

    for (int st = wave; st < NTILE; st += 8) {     // 8 uniform iterations/wave
        const int scol = (st << 4) + ln;

        if (st + 8 < NTILE)  // wave-uniform; hint the next gt_dT tile
            __builtin_prefetch(&gt4[((size_t)b * T_ + (t0 + half * 8)) * T_ + scol + 128], 0, 0);

        // w tile, matching C-matrix lane layout; read-once stream -> TH=NT
        float wv[8];
#pragma unroll
        for (int j = 0; j < 8; ++j) {
            const int tr = t0 + j + half * 8;
            const vf4 dt = __builtin_nontemporal_load(
                &gt4[((size_t)b * T_ + tr) * T_ + scol]);
            const float e = CYX * (dt.x * dt.x + dt.y * dt.y) +
                            CHW * (dt.z * dt.z + dt.w * dt.w);
            wv[j] = exp2f(e);                       // v_exp_f32
        }

        // Gram tile G = Zt * Zs^T : 8 WMMAs over K = 256
        v8f c = {};
        const size_t browC = ((size_t)b * T_ + scol) * 32;
#pragma unroll
        for (int ks = 0; ks < 8; ++ks) {
            const v16bf bfrag = load_frag(zb4, browC, ks, half);
            c = __builtin_amdgcn_wmma_f32_16x16x32_bf16(
                    false, afrag[ks], false, bfrag,
                    (short)0, c, false, false);
        }

        const float z2s = z2[b * T_ + scol];
#pragma unroll
        for (int j = 0; j < 8; ++j) {
            accN = fmaf(wv[j], z2t[j] + z2s - 2.0f * c[j], accN);
            accD += wv[j];
        }
    }

    __shared__ float sN[256];
    __shared__ float sD[256];
    sN[tid] = accN;
    sD[tid] = accD;
    __syncthreads();
#pragma unroll
    for (int off = 128; off > 0; off >>= 1) {
        if (tid < off) { sN[tid] += sN[tid + off]; sD[tid] += sD[tid + off]; }
        __syncthreads();
    }
    if (tid == 0) {
        atomicAdd(&accum[b], sN[0]);
        atomicAdd(&accum[B_ + b], sD[0]);
    }
}

// ---------------------------------------------------------------------------
// Final: mean_b num[b] / max(den[b], 1e-6)
// ---------------------------------------------------------------------------
__global__ __launch_bounds__(32)
void final_kernel(const float* __restrict__ accum, float* __restrict__ out) {
    __shared__ float r[B_];
    const int t = threadIdx.x;
    if (t < B_) {
        const float den = fmaxf(accum[B_ + t], 1e-6f);
        r[t] = accum[t] / den;
    }
    __syncthreads();
    if (t == 0) {
        float s = 0.0f;
#pragma unroll
        for (int i = 0; i < B_; ++i) s += r[i];
        out[0] = s * (1.0f / (float)B_);
    }
}

// ---------------------------------------------------------------------------
extern "C" void kernel_launch(void* const* d_in, const int* in_sizes, int n_in,
                              void* d_out, int out_size, void* d_ws, size_t ws_size,
                              hipStream_t stream) {
    (void)in_sizes; (void)n_in; (void)out_size; (void)ws_size;
    const float* z  = (const float*)d_in[0];     // (B,T,D) f32
    const float* gt = (const float*)d_in[1];     // (B,T,T,4) f32
    float* out = (float*)d_out;

    char* ws = (char*)d_ws;
    unsigned int* zbu = (unsigned int*)ws;                              // 8 MB bf16 z
    float* z2    = (float*)(ws + (size_t)8 * 1024 * 1024);              // 64 KB
    float* accum = (float*)(ws + (size_t)8 * 1024 * 1024 + 65536);      // 32 f32

    hipMemsetAsync(accum, 0, 2 * B_ * sizeof(float), stream);

    prep_kernel<<<B_ * T_, 128, 0, stream>>>(z, zbu, z2);
    lap_energy_kernel<<<B_ * NTILE, 256, 0, stream>>>(
        (const vf4*)gt, (const uint4*)zbu, z2, accum);
    final_kernel<<<1, 32, 0, stream>>>(accum, out);
}
